// PLEBasicLayer_35330400977385
// MI455X (gfx1250) — matrix operand order repaired and checked by reading, not compiled, not run on hardware
//
#include <hip/hip_runtime.h>
#include <math.h>

typedef __attribute__((ext_vector_type(16))) __bf16 v16bf;
typedef __attribute__((ext_vector_type(8)))  float  v8f;
typedef __attribute__((ext_vector_type(4)))  unsigned v4u;
typedef __attribute__((ext_vector_type(8)))  unsigned v8u;

#define O_     3
#define CELL_  5
#define IDIM_  4
#define ODIM_  64
#define KW_    15
#define BS_    16
#define L_     2048
#define NB_    18
#define CRUC_  38

// ---------------- workspace layout (floats) ----------------
constexpr size_t HSZ       = (size_t)NB_*BS_*ODIM_*L_;     // 151MB: hpre -> h -> hidden (in place)
constexpr size_t OFF_H     = 0;
constexpr size_t OFF_BNSUM = OFF_H + HSZ;                  // [NB][ODIM]
constexpr size_t OFF_BNSSQ = OFF_BNSUM + NB_*ODIM_;
constexpr size_t OFF_ATTST = OFF_BNSSQ + NB_*ODIM_;        // [NB][2 var][sum,ssq]
constexpr size_t OFF_ZEND  = OFF_ATTST + NB_*4;            // zeroed region end
constexpr size_t OFF_BNSC  = OFF_ZEND;                     // [NB][ODIM] bn scale
constexpr size_t OFF_BNSH  = OFF_BNSC + NB_*ODIM_;         // bn shift
constexpr size_t OFF_AVG   = OFF_BNSH + NB_*ODIM_;         // [NB][BS][ODIM] (sum over L)
constexpr size_t OFF_MAX   = OFF_AVG + NB_*BS_*ODIM_;
constexpr size_t OFF_Y     = OFF_MAX + NB_*BS_*ODIM_;
constexpr size_t OFF_MASK  = OFF_Y + NB_*BS_*ODIM_;
constexpr size_t ZPLANE    = (size_t)NB_*BS_*L_;
constexpr size_t OFF_ZCFMX = OFF_MASK + NB_*BS_*ODIM_;
constexpr size_t OFF_ZCFMN = OFF_ZCFMX + ZPLANE;
constexpr size_t OFF_ZSFMX = OFF_ZCFMN + ZPLANE;
constexpr size_t OFF_ZSFMN = OFF_ZSFMX + ZPLANE;
constexpr size_t OFF_AACF  = OFF_ZSFMN + ZPLANE;
constexpr size_t OFF_AASF  = OFF_AACF + ZPLANE;
constexpr size_t OFF_ATTSC = OFF_AASF + ZPLANE;            // [NB][2]
constexpr size_t OFF_ATTSH = OFF_ATTSC + NB_*2;
constexpr size_t OFF_XMEAN = OFF_ATTSH + NB_*2;            // [5][3][16][4]
constexpr size_t OFF_G0    = OFF_XMEAN + 5*O_*BS_*IDIM_;   // [16][18]
constexpr size_t OFF_GS    = OFF_G0 + BS_*18;              // [5][16][6]

constexpr size_t OSZ       = (size_t)O_*BS_*ODIM_*L_;      // per-output tensor

__device__ __forceinline__ float sigmoidf_(float x) { return 1.f/(1.f + __expf(-x)); }

// ---------------- K0: zero the atomic-accumulated stats ----------------
__global__ void k0_zero(float* p, int n) {
    int i = blockIdx.x*256 + threadIdx.x;
    if (i < n) p[i] = 0.f;
}

// ---------------- K1: conv1d as implicit GEMM via WMMA bf16 ----------------
// Activation plane staged to LDS by the Tensor Data Mover (TDM, TENSORcnt);
// weight block staged by async global->LDS copies (ASYNCcnt). BN stats fused.
__global__ __launch_bounds__(256) void k1_conv_wmma(
    const float* __restrict__ omics, const float* __restrict__ convw,
    const float* __restrict__ convb, float* __restrict__ ws)
{
    __shared__ float  Axf[IDIM_*L_];           // 32KB raw fp32 activations (TDM dest)
    __shared__ float  Wst[ODIM_*IDIM_*KW_];    // 15KB raw fp32 weights (async dest)
    __shared__ __bf16 Bsh[64][68];             // weights, kk-major bf16, padded
    __shared__ float  ssum[ODIM_], sssq[ODIM_];

    const int tid = threadIdx.x;
    const int bb  = blockIdx.x >> 4;           // ham block
    const int n   = blockIdx.x & 15;           // batch
    const int k6  = bb / 3;
    const int o   = bb - 3*k6;
    const int page = (k6 == 0) ? 0 : (k6 - 1); // src = [0,0,1,2,3,4]

    const float* xb   = omics + ((size_t)(page*O_ + o)*BS_ + n)*IDIM_*L_;
    const float* wsrc = convw + (size_t)bb*ODIM_*IDIM_*KW_;

    // --- TDM: one 2-D tensor_load_to_lds for the whole 4x2048 fp32 plane ---
    if (tid < 32) {
        unsigned long long ga = (unsigned long long)(uintptr_t)xb;
        unsigned ldsa = (unsigned)(uintptr_t)&Axf[0];
        v4u g0;
        g0[0] = 1u;                                             // count=1, user mode
        g0[1] = ldsa;                                           // lds_addr (bytes)
        g0[2] = (unsigned)(ga & 0xffffffffu);                   // global_addr[31:0]
        g0[3] = (unsigned)((ga >> 32) & 0x01ffffffu) | (2u<<30);// global_addr[56:32] | type=2
        v8u g1;
        g1[0] = 0x00020000u;                  // workgroup_mask=0, data_size=2 (4B)
        g1[1] = (unsigned)(L_ & 0xffff) << 16;        // tensor_dim0[15:0]
        g1[2] = (unsigned)(L_ >> 16) | (IDIM_ << 16); // tensor_dim0[31:16] | tensor_dim1[15:0]
        g1[3] = (unsigned)(L_ << 16);                 // tensor_dim1[31:16]=0 | tile_dim0
        g1[4] = IDIM_;                                // tile_dim1 | tile_dim2=0
        g1[5] = L_;                                   // tensor_dim0_stride[31:0]
        g1[6] = 0u;                                   // stride hi | tensor_dim1_stride lo
        g1[7] = 0u;
        asm volatile("tensor_load_to_lds %0, %1" :: "s"(g0), "s"(g1) : "memory");
        __builtin_amdgcn_s_wait_tensorcnt(0);
    }

    // --- async global->LDS (no VGPR landing) for the raw weight block ---
    for (int idx = tid; idx < ODIM_*IDIM_*KW_/4; idx += 256) {
        unsigned ldso = (unsigned)(uintptr_t)&Wst[idx*4];
        unsigned long long ga2 = (unsigned long long)(uintptr_t)(wsrc + idx*4);
        asm volatile("global_load_async_to_lds_b128 %0, %1, off"
                     :: "v"(ldso), "v"(ga2) : "memory");
    }
    asm volatile("s_wait_asynccnt 0x0" ::: "memory");
    if (tid < ODIM_) { ssum[tid] = 0.f; sssq[tid] = 0.f; }
    __syncthreads();

    // rearrange weights: [c][i][t] fp32 -> Bsh[kk][c] bf16 (kk = i*KW+t, zero-pad to 64)
    for (int idx = tid; idx < 64*ODIM_; idx += 256) {
        int kk = idx >> 6, c = idx & 63;
        float w = 0.f;
        if (kk < IDIM_*KW_) {
            int i = kk / KW_, t = kk - i*KW_;
            w = Wst[(c*IDIM_ + i)*KW_ + t];
        }
        Bsh[kk][c] = (__bf16)w;
    }
    __syncthreads();

    const int lane = tid & 31, wid = tid >> 5;
    const int m = lane & 15, hi = lane >> 4;

    // B fragments: 32x16 bf16 per (ctile, ksplit); loop-invariant
    v16bf bfrag[4][2];
    #pragma unroll
    for (int ct = 0; ct < 4; ++ct) {
        int cc = ct*16 + m;
        #pragma unroll
        for (int s = 0; s < 2; ++s) {
            int kb = s*32 + hi*16;
            v16bf bf;
            #pragma unroll
            for (int j = 0; j < 8; ++j) {
                bf[2*j]   = Bsh[kb + 2*j][cc];
                bf[2*j+1] = Bsh[kb + 2*j + 1][cc];
            }
            bfrag[ct][s] = bf;
        }
    }
    float cbv[4];
    #pragma unroll
    for (int ct = 0; ct < 4; ++ct) cbv[ct] = convb[bb*ODIM_ + ct*16 + m];

    float lsum[4] = {0,0,0,0}, lssq[4] = {0,0,0,0};
    const size_t rowbase = (size_t)(bb*BS_ + n)*ODIM_;
    const v8f vz = {0.f,0.f,0.f,0.f,0.f,0.f,0.f,0.f};

    for (int tile = wid; tile < L_/16; tile += 8) {
        const int l0 = tile*16, lpos = l0 + m;
        // A fragments: 16x32 bf16 per ksplit, contiguous K layout (k = 16*hi + element), same as B
        v16bf afr[2];
        #pragma unroll
        for (int s = 0; s < 2; ++s) {
            v16bf a;
            #pragma unroll
            for (int half = 0; half < 2; ++half) {
                int kb = s*32 + hi*16 + half*8;
                #pragma unroll
                for (int e = 0; e < 8; ++e) {
                    int kk = kb + e;
                    __bf16 v = (__bf16)0.f;
                    if (kk < IDIM_*KW_) {
                        int i = kk / KW_, t = kk - i*KW_;
                        int p = lpos + t - (KW_/2);
                        if (p >= 0 && p < L_) v = (__bf16)Axf[i*L_ + p];
                    }
                    a[half*8 + e] = v;
                }
            }
            afr[s] = a;
        }
        #pragma unroll
        for (int ct = 0; ct < 4; ++ct) {
            v8f acc = vz;
            acc = __builtin_amdgcn_wmma_f32_16x16x32_bf16(false, afr[0], false, bfrag[ct][0], (short)0, acc, false, false);
            acc = __builtin_amdgcn_wmma_f32_16x16x32_bf16(false, afr[1], false, bfrag[ct][1], (short)0, acc, false, false);
            // epilogue: lane owns (c = ct*16+m), 8 consecutive l values
            const int c = ct*16 + m;
            const size_t base = (rowbase + c)*L_ + l0 + hi*8;
            float4 v0, v1;
            v0.x = acc[0]+cbv[ct]; v0.y = acc[1]+cbv[ct]; v0.z = acc[2]+cbv[ct]; v0.w = acc[3]+cbv[ct];
            v1.x = acc[4]+cbv[ct]; v1.y = acc[5]+cbv[ct]; v1.z = acc[6]+cbv[ct]; v1.w = acc[7]+cbv[ct];
            *(float4*)(ws + base)     = v0;
            *(float4*)(ws + base + 4) = v1;
            lsum[ct] += v0.x+v0.y+v0.z+v0.w + v1.x+v1.y+v1.z+v1.w;
            lssq[ct] += v0.x*v0.x+v0.y*v0.y+v0.z*v0.z+v0.w*v0.w
                      + v1.x*v1.x+v1.y*v1.y+v1.z*v1.z+v1.w*v1.w;
        }
    }
    #pragma unroll
    for (int ct = 0; ct < 4; ++ct) {
        atomicAdd(&ssum[ct*16+m], lsum[ct]);
        atomicAdd(&sssq[ct*16+m], lssq[ct]);
    }
    __syncthreads();
    if (tid < ODIM_) {
        atomicAdd(&ws[OFF_BNSUM + bb*ODIM_ + tid], ssum[tid]);
        atomicAdd(&ws[OFF_BNSSQ + bb*ODIM_ + tid], sssq[tid]);
    }
}

// ---------------- K2: finalize BN scale/shift ----------------
__global__ void k2_bnfin(const float* __restrict__ bn_g, const float* __restrict__ bn_b, float* ws) {
    int i = blockIdx.x*256 + threadIdx.x;
    if (i >= NB_*ODIM_) return;
    const float inv = 1.f/(float)(BS_*L_);
    float mean = ws[OFF_BNSUM+i]*inv;
    float var  = ws[OFF_BNSSQ+i]*inv - mean*mean;
    float sc = bn_g[i]*rsqrtf(var + 1e-5f);
    ws[OFF_BNSC+i] = sc;
    ws[OFF_BNSH+i] = bn_b[i] - mean*sc;
}

// ---------------- K3: BN+ReLU in place + sum/max pooling over L ----------------
__global__ __launch_bounds__(256) void k3_bn_pool(float* __restrict__ ws) {
    int row  = blockIdx.x*8 + (threadIdx.x >> 5);   // (bb,n,c) row, < 18432
    int lane = threadIdx.x & 31;
    int bb = row >> 10, c = row & 63;
    float sc = ws[OFF_BNSC + bb*ODIM_ + c];
    float sh = ws[OFF_BNSH + bb*ODIM_ + c];
    float4* hp = (float4*)(ws + (size_t)row*L_);
    float s = 0.f, mx = 0.f;
    for (int t = 0; t < 16; ++t) {
        float4 v = hp[t*32 + lane];
        v.x = fmaxf(fmaf(v.x, sc, sh), 0.f); v.y = fmaxf(fmaf(v.y, sc, sh), 0.f);
        v.z = fmaxf(fmaf(v.z, sc, sh), 0.f); v.w = fmaxf(fmaf(v.w, sc, sh), 0.f);
        hp[t*32 + lane] = v;
        s += v.x+v.y+v.z+v.w;
        mx = fmaxf(mx, fmaxf(fmaxf(v.x, v.y), fmaxf(v.z, v.w)));
    }
    for (int d = 16; d; d >>= 1) { s += __shfl_xor(s, d, 32); mx = fmaxf(mx, __shfl_xor(mx, d, 32)); }
    if (!lane) { ws[OFF_AVG+row] = s; ws[OFF_MAX+row] = mx; }
}

// ---------------- K4: channel attention y + top-38 mask (one wave per (bb,n))
__global__ __launch_bounds__(256) void k4_changate(
    const float* __restrict__ ca_a, const float* __restrict__ ca_be,
    const float* __restrict__ ca_w, const float* __restrict__ ca_b, float* __restrict__ ws)
{
    __shared__ float fL[8][ODIM_];
    __shared__ float yL[8][ODIM_];
    int wid = threadIdx.x >> 5, lane = threadIdx.x & 31;
    int idx = blockIdx.x*8 + wid;       // < 288
    int b = idx >> 4;
    float a = ca_a[b] + 0.5f, be = ca_be[b] + 0.5f;
    for (int c = lane; c < ODIM_; c += 32) {
        float av = ws[OFF_AVG + idx*ODIM_ + c]*(1.f/L_);
        float mx = ws[OFF_MAX + idx*ODIM_ + c];
        fL[wid][c] = a*av + be*mx;
    }
    __syncthreads();
    float w0 = ca_w[b*3], w1 = ca_w[b*3+1], w2 = ca_w[b*3+2], cb = ca_b[b];
    for (int c = lane; c < ODIM_; c += 32) {
        float lft = c ? fL[wid][c-1] : 0.f;
        float rgt = (c < ODIM_-1) ? fL[wid][c+1] : 0.f;
        float y = sigmoidf_(fmaf(w0, lft, fmaf(w1, fL[wid][c], fmaf(w2, rgt, cb))));
        yL[wid][c] = y;
        ws[OFF_Y + idx*ODIM_ + c] = y;
    }
    __syncthreads();
    for (int c = lane; c < ODIM_; c += 32) {
        float yc = yL[wid][c];
        int r = 0;
        for (int j = 0; j < ODIM_; ++j) {
            float yj = yL[wid][j];
            r += (yj > yc) || (yj == yc && j < c);
        }
        ws[OFF_MASK + idx*ODIM_ + c] = (r < CRUC_) ? 1.f : 0.f;
    }
}

// ---------------- K5: channel max/mean of cf/sf per (bb,n,l) ----------------
__global__ __launch_bounds__(256) void k5_zcalc(float* __restrict__ ws) {
    __shared__ float yL[ODIM_], mL[ODIM_];
    int bn = blockIdx.x >> 3;
    int l  = ((blockIdx.x & 7) << 8) + threadIdx.x;
    if (threadIdx.x < ODIM_) {
        yL[threadIdx.x] = ws[OFF_Y    + bn*ODIM_ + threadIdx.x];
        mL[threadIdx.x] = ws[OFF_MASK + bn*ODIM_ + threadIdx.x];
    }
    __syncthreads();
    const float* h = ws + (size_t)bn*ODIM_*L_;
    float cfm = 0.f, cfs = 0.f, sfm = 0.f, sfs = 0.f;
    for (int c = 0; c < ODIM_; ++c) {
        if (c + 4 < ODIM_) __builtin_prefetch(&h[(size_t)(c+4)*L_ + l], 0, 0);
        float s = h[(size_t)c*L_ + l] * yL[c];
        if (mL[c] > 0.f) { cfm = fmaxf(cfm, s); cfs += s; }
        else             { sfm = fmaxf(sfm, s); sfs += s; }
    }
    size_t zi = (size_t)bn*L_ + l;
    ws[OFF_ZCFMX+zi] = cfm;
    ws[OFF_ZCFMN+zi] = cfs * (1.f/(float)CRUC_);            // (sum/64)*(64/38)
    ws[OFF_ZSFMX+zi] = sfm;
    ws[OFF_ZSFMN+zi] = sfs * (1.f/(float)(ODIM_-CRUC_));    // (sum/64)*(64/26)
}

// ---------------- K6: 7-tap spatial conv + attention BN stats ----------------
__global__ __launch_bounds__(256) void k6_saconv(const float* __restrict__ sa_w, float* __restrict__ ws) {
    int bn = blockIdx.x >> 3;
    int l  = ((blockIdx.x & 7) << 8) + threadIdx.x;
    int b  = bn >> 4;
    float wmx[7], wmn[7];
    #pragma unroll
    for (int t = 0; t < 7; ++t) { wmx[t] = sa_w[b*14 + t]; wmn[t] = sa_w[b*14 + 7 + t]; }
    const float* zcx = ws + OFF_ZCFMX + (size_t)bn*L_;
    const float* zcn = ws + OFF_ZCFMN + (size_t)bn*L_;
    const float* zsx = ws + OFF_ZSFMX + (size_t)bn*L_;
    const float* zsn = ws + OFF_ZSFMN + (size_t)bn*L_;
    float acf = 0.f, asf = 0.f;
    #pragma unroll
    for (int t = 0; t < 7; ++t) {
        int p = l + t - 3;
        if (p >= 0 && p < L_) {
            acf += wmx[t]*zcx[p] + wmn[t]*zcn[p];
            asf += wmx[t]*zsx[p] + wmn[t]*zsn[p];
        }
    }
    ws[OFF_AACF + (size_t)bn*L_ + l] = acf;
    ws[OFF_AASF + (size_t)bn*L_ + l] = asf;
    float v[4] = {acf, acf*acf, asf, asf*asf};
    int lane = threadIdx.x & 31;
    for (int d = 16; d; d >>= 1)
        for (int q = 0; q < 4; ++q) v[q] += __shfl_xor(v[q], d, 32);
    __shared__ float red[4];
    if (threadIdx.x < 4) red[threadIdx.x] = 0.f;
    __syncthreads();
    if (!lane) for (int q = 0; q < 4; ++q) atomicAdd(&red[q], v[q]);
    __syncthreads();
    if (threadIdx.x < 4) atomicAdd(&ws[OFF_ATTST + b*4 + threadIdx.x], red[threadIdx.x]);
}

// ---------------- K7: attention BN finalize ----------------
__global__ void k7_attfin(const float* __restrict__ sa_g, const float* __restrict__ sa_b, float* ws) {
    int i = threadIdx.x;
    if (i >= NB_*2) return;
    int b = i >> 1, var = i & 1;
    const float inv = 1.f/(float)(BS_*L_);
    float mean = ws[OFF_ATTST + b*4 + var*2]*inv;
    float ssq  = ws[OFF_ATTST + b*4 + var*2 + 1]*inv;
    float sc = sa_g[b]*rsqrtf(ssq - mean*mean + 1e-5f);
    ws[OFF_ATTSC + i] = sc;
    ws[OFF_ATTSH + i] = sa_b[b] - mean*sc;
}

// ---------------- K8: hidden = h * y * sigmoid(relu(bn(aa))), in place -----
__global__ __launch_bounds__(256) void k8_hidden(float* __restrict__ ws) {
    size_t idx4 = (size_t)blockIdx.x*256 + threadIdx.x;   // < HSZ/4
    int l4  = (int)(idx4 & 511);
    int c   = (int)((idx4 >> 9) & 63);
    size_t bn = idx4 >> 15;
    float y  = ws[OFF_Y    + bn*ODIM_ + c];
    float mk = ws[OFF_MASK + bn*ODIM_ + c];
    int b = (int)(bn >> 4);
    int var = (mk > 0.f) ? 0 : 1;
    float sc = ws[OFF_ATTSC + b*2 + var], sh = ws[OFF_ATTSH + b*2 + var];
    const float* aa = ws + ((var == 0) ? OFF_AACF : OFF_AASF) + bn*L_;
    float4 a4 = ((const float4*)aa)[l4];
    float4 h4 = ((float4*)ws)[idx4];
    h4.x *= y * sigmoidf_(fmaxf(fmaf(a4.x, sc, sh), 0.f));
    h4.y *= y * sigmoidf_(fmaxf(fmaf(a4.y, sc, sh), 0.f));
    h4.z *= y * sigmoidf_(fmaxf(fmaf(a4.z, sc, sh), 0.f));
    h4.w *= y * sigmoidf_(fmaxf(fmaf(a4.w, sc, sh), 0.f));
    ((float4*)ws)[idx4] = h4;
}

// ---------------- K9: omics mean over L (one wave per row) ----------------
__global__ __launch_bounds__(256) void k9_xmean(const float* __restrict__ omics, float* __restrict__ ws) {
    int row  = blockIdx.x*8 + (threadIdx.x >> 5);   // < 960
    int lane = threadIdx.x & 31;
    const float4* p = (const float4*)(omics + (size_t)row*L_);
    float s = 0.f;
    for (int t = 0; t < 16; ++t) { float4 v = p[t*32 + lane]; s += v.x+v.y+v.z+v.w; }
    for (int d = 16; d; d >>= 1) s += __shfl_xor(s, d, 32);
    if (!lane) ws[OFF_XMEAN + row] = s*(1.f/L_);
}

// ---------------- K10: gate matmuls + softmax (single block) ----------------
__global__ __launch_bounds__(256) void k10_gates(
    const float* __restrict__ gsw, const float* __restrict__ gsb,
    const float* __restrict__ gpw, const float* __restrict__ gpb, float* __restrict__ ws)
{
    __shared__ float xm[5*O_*BS_*IDIM_];
    __shared__ float lg0[BS_*18];
    __shared__ float lgs[5*BS_*6];
    int tid = threadIdx.x;
    for (int i = tid; i < 5*O_*BS_*IDIM_; i += 256) xm[i] = ws[OFF_XMEAN + i];
    __syncthreads();
    for (int t = tid; t < BS_*18; t += 256) {
        int n = t/18, j = t - 18*n;
        float s = gsb[j];
        #pragma unroll
        for (int oi = 0; oi < 12; ++oi) {
            int o = oi >> 2, i = oi & 3;
            s += xm[((0*O_ + o)*BS_ + n)*IDIM_ + i] * gsw[j*12 + oi];
        }
        lg0[t] = s;
    }
    for (int t = tid; t < 5*BS_*6; t += 256) {
        int p = t/96, rem = t - 96*p, n = rem/6, j = rem - 6*n;
        float s = gpb[p*6 + j];
        #pragma unroll
        for (int oi = 0; oi < 12; ++oi) {
            int o = oi >> 2, i = oi & 3;
            s += xm[((p*O_ + o)*BS_ + n)*IDIM_ + i] * gpw[(p*6 + j)*12 + oi];
        }
        lgs[t] = s;
    }
    __syncthreads();
    if (tid < BS_) {
        float mx = -1e30f;
        for (int j = 0; j < 18; ++j) mx = fmaxf(mx, lg0[tid*18 + j]);
        float sm = 0.f;
        for (int j = 0; j < 18; ++j) sm += __expf(lg0[tid*18 + j] - mx);
        float inv = 1.f/sm;
        for (int j = 0; j < 18; ++j) ws[OFF_G0 + tid*18 + j] = __expf(lg0[tid*18 + j] - mx)*inv;
    }
    if (tid >= 64 && tid < 64 + 5*BS_) {
        int q = tid - 64, p = q/16, n = q & 15;
        float mx = -1e30f;
        for (int j = 0; j < 6; ++j) mx = fmaxf(mx, lgs[(p*16+n)*6 + j]);
        float sm = 0.f;
        for (int j = 0; j < 6; ++j) sm += __expf(lgs[(p*16+n)*6 + j] - mx);
        float inv = 1.f/sm;
        for (int j = 0; j < 6; ++j) ws[OFF_GS + (p*16+n)*6 + j] = __expf(lgs[(p*16+n)*6 + j] - mx)*inv;
    }
}

// ---------------- K11: gated combine, read 6 hidden planes once, write 6 outputs
__global__ __launch_bounds__(256) void k11_combine(float* __restrict__ out, const float* __restrict__ ws) {
    size_t idx4 = (size_t)blockIdx.x*256 + threadIdx.x;   // < OSZ/4
    int l4 = (int)(idx4 & 511);
    int c  = (int)((idx4 >> 9) & 63);
    int n  = (int)((idx4 >> 15) & 15);
    int o  = (int)(idx4 >> 19);
    float4 hk[6];
    #pragma unroll
    for (int k = 0; k < 6; ++k)
        hk[k] = ((const float4*)ws)[((size_t)((k*O_ + o)*BS_ + n)*ODIM_ + c)*(L_/4) + l4];
    float4 acc; acc.x = acc.y = acc.z = acc.w = 0.f;
    #pragma unroll
    for (int k = 0; k < 6; ++k) {
        float g = ws[OFF_G0 + n*18 + o*6 + k];
        acc.x += g*hk[k].x; acc.y += g*hk[k].y; acc.z += g*hk[k].z; acc.w += g*hk[k].w;
    }
    const float inv6 = 1.f/6.f;
    float4 r0; r0.x = acc.x*inv6; r0.y = acc.y*inv6; r0.z = acc.z*inv6; r0.w = acc.w*inv6;
    ((float4*)out)[idx4] = r0;
    #pragma unroll
    for (int p = 0; p < CELL_; ++p) {
        float g0 = ws[OFF_GS + (p*16+n)*6 + o*2 + 0];
        float g1 = ws[OFF_GS + (p*16+n)*6 + o*2 + 1];
        float4 r;
        r.x = 0.5f*(g0*hk[0].x + g1*hk[p+1].x);
        r.y = 0.5f*(g0*hk[0].y + g1*hk[p+1].y);
        r.z = 0.5f*(g0*hk[0].z + g1*hk[p+1].z);
        r.w = 0.5f*(g0*hk[0].w + g1*hk[p+1].w);
        ((float4*)out)[(size_t)(p+1)*(OSZ/4) + idx4] = r;
    }
}

extern "C" void kernel_launch(void* const* d_in, const int* in_sizes, int n_in,
                              void* d_out, int out_size, void* d_ws, size_t ws_size,
                              hipStream_t stream) {
    const float* omics  = (const float*)d_in[0];
    const float* conv_w = (const float*)d_in[1];
    const float* conv_b = (const float*)d_in[2];
    const float* bn_g   = (const float*)d_in[3];
    const float* bn_b   = (const float*)d_in[4];
    const float* ca_a   = (const float*)d_in[5];
    const float* ca_be  = (const float*)d_in[6];
    const float* ca_w   = (const float*)d_in[7];
    const float* ca_b   = (const float*)d_in[8];
    const float* sa_w   = (const float*)d_in[9];
    const float* sa_g   = (const float*)d_in[10];
    const float* sa_b   = (const float*)d_in[11];
    const float* gsw    = (const float*)d_in[12];
    const float* gsb    = (const float*)d_in[13];
    const float* gpw    = (const float*)d_in[14];
    const float* gpb    = (const float*)d_in[15];
    float* ws  = (float*)d_ws;
    float* out = (float*)d_out;

    const int nz = (int)(OFF_ZEND - OFF_BNSUM);
    k0_zero<<<(nz + 255)/256, 256, 0, stream>>>(ws + OFF_BNSUM, nz);
    k1_conv_wmma<<<NB_*BS_, 256, 0, stream>>>(omics, conv_w, conv_b, ws);
    k2_bnfin<<<(NB_*ODIM_ + 255)/256, 256, 0, stream>>>(bn_g, bn_b, ws);
    k3_bn_pool<<<NB_*BS_*ODIM_/8, 256, 0, stream>>>(ws);
    k4_changate<<<NB_*BS_/8, 256, 0, stream>>>(ca_a, ca_be, ca_w, ca_b, ws);
    k5_zcalc<<<NB_*BS_*(L_/256), 256, 0, stream>>>(ws);
    k6_saconv<<<NB_*BS_*(L_/256), 256, 0, stream>>>(sa_w, ws);
    k7_attfin<<<1, 64, 0, stream>>>(sa_g, sa_b, ws);
    k8_hidden<<<(unsigned)(HSZ/4/256), 256, 0, stream>>>(ws);
    k9_xmean<<<5*O_*BS_*IDIM_/8, 256, 0, stream>>>(omics, ws);
    k10_gates<<<1, 256, 0, stream>>>(gsw, gsb, gpw, gpb, ws);
    k11_combine<<<(unsigned)(OSZ/4/256), 256, 0, stream>>>(out, ws);
}
